// GPT_48825188221653
// MI455X (gfx1250) — matrix operand order repaired and checked by reading, not compiled, not run on hardware
//
#include <hip/hip_runtime.h>
#include <hip/hip_bf16.h>

// ---------------- problem constants ----------------
constexpr int kB = 4;          // batch
constexpr int kT = 256;        // timesteps
constexpr int kS = 3 * kT;     // token sequence = 768
constexpr int kC = 1024;       // embed dim
constexpr int kH = 16;         // heads
constexpr int kD = 64;         // head dim
constexpr int kF = 4096;       // ffn dim
constexpr int kL = 6;          // layers
constexpr int kA = 80;         // n actions
constexpr int kM = kB * kS;    // 3072 GEMM rows

// ---------------- CDNA5 WMMA types ----------------
typedef __bf16 bf16_t;
typedef bf16_t v16bf __attribute__((ext_vector_type(16)));
typedef float  v8f   __attribute__((ext_vector_type(8)));
typedef int    v4i_t __attribute__((vector_size(16)));   // matches async-LDS builtin proto

#ifndef __has_builtin
#define __has_builtin(x) 0
#endif
#if __has_builtin(__builtin_amdgcn_global_load_async_to_lds_b128)
#define USE_ASYNC_LDS 1
#else
#define USE_ASYNC_LDS 0
#endif

__device__ __forceinline__ unsigned short f2bf(float f) {
  unsigned int u = __float_as_uint(f);
  u += 0x7FFFu + ((u >> 16) & 1u);          // round-to-nearest-even
  return (unsigned short)(u >> 16);
}

union FragU {
  uint4 q[2];
  v16bf v;
  unsigned short us[16];
};

__device__ __forceinline__ v16bf load2x16(const unsigned short* p0,
                                          const unsigned short* p1) {
  FragU u;
  u.q[0] = *(const uint4*)p0;
  u.q[1] = *(const uint4*)p1;
  return u.v;
}

__device__ __forceinline__ v8f wmma_bf16(v16bf a, v16bf b, v8f c) {
  // D = A(16x32 bf16) * B(32x16 bf16) + C(16x16 f32)
  return __builtin_amdgcn_wmma_f32_16x16x32_bf16(
      /*neg_a=*/false, a, /*neg_b=*/false, b,
      /*c_mod=*/(short)0, c, /*reuse_a=*/false, /*reuse_b=*/false);
}

// 16-byte global -> LDS copy (async on gfx1250, sync register fallback)
__device__ __forceinline__ void cp16_g2l(unsigned short* lds_dst,
                                         const unsigned short* gsrc) {
#if USE_ASYNC_LDS
  __builtin_amdgcn_global_load_async_to_lds_b128(
      (__attribute__((address_space(1))) v4i_t*)gsrc,
      (__attribute__((address_space(3))) v4i_t*)lds_dst,
      /*imm_offset=*/0, /*cpol=*/0);
#else
  *(uint4*)lds_dst = *(const uint4*)gsrc;
#endif
}

__device__ __forceinline__ void wait_async() {
#if USE_ASYNC_LDS
#if __has_builtin(__builtin_amdgcn_s_wait_asynccnt)
  __builtin_amdgcn_s_wait_asynccnt(0);
#else
  asm volatile("s_wait_asynccnt 0x0" ::: "memory");
#endif
#endif
}

// ---------------- weight cast + transpose: dst[n*K+k] = bf16(src[k*N+n]) ----------------
__global__ void castT_k(const float* __restrict__ src, unsigned short* __restrict__ dst,
                        int K, int N, int Npad) {
  size_t idx = (size_t)blockIdx.x * blockDim.x + threadIdx.x;
  size_t tot = (size_t)Npad * K;
  if (idx >= tot) return;
  int n = (int)(idx / K);
  int k = (int)(idx % K);
  float v = (n < N) ? src[(size_t)k * N + n] : 0.0f;
  dst[idx] = f2bf(v);
}

// ---------------- embedding front-end: CNN + meta MLP + token interleave ----------------
__global__ __launch_bounds__(256) void embed_k(
    const float* __restrict__ states, const float* __restrict__ rtgs,
    const float* __restrict__ meta,
    const float* __restrict__ c1w, const float* __restrict__ c1b,
    const float* __restrict__ c2w, const float* __restrict__ c2b,
    const float* __restrict__ c3w, const float* __restrict__ c3b,
    const float* __restrict__ encw, const float* __restrict__ encb,
    const float* __restrict__ mw1, const float* __restrict__ mb1,
    const float* __restrict__ mw2, const float* __restrict__ mb2,
    const float* __restrict__ retw, const float* __restrict__ retb,
    const float* __restrict__ actemb, const float* __restrict__ posemb,
    const float* __restrict__ gposemb,
    const int* __restrict__ actions, const int* __restrict__ timesteps,
    float* __restrict__ x) {
  int bt = blockIdx.x;
  int b = bt / kT, t = bt % kT;
  const float* s = states + (size_t)bt * 640;   // (8ch, 8h, 10w)

  __shared__ float h1[16][2][3];
  __shared__ float h2[32];
  __shared__ float feats[16];
  __shared__ float mh[32];
  __shared__ float me[4];

  int tid = threadIdx.x;
  // conv1: (16,8,8,8) s2 p1 -> (16,2,3)
  if (tid < 96) {
    int oc = tid / 6, r = tid % 6, oy = r / 3, ox = r % 3;
    float acc = c1b[oc];
    for (int ic = 0; ic < 8; ic++)
      for (int ky = 0; ky < 8; ky++) {
        int iy = oy * 2 - 1 + ky;
        if (iy < 0 || iy >= 8) continue;
        for (int kx = 0; kx < 8; kx++) {
          int ix = ox * 2 - 1 + kx;
          if (ix < 0 || ix >= 10) continue;
          acc += s[ic * 80 + iy * 10 + ix] * c1w[((oc * 8 + ic) * 8 + ky) * 8 + kx];
        }
      }
    h1[oc][oy][ox] = fmaxf(acc, 0.0f);
  }
  __syncthreads();
  // conv2: (32,16,4,4) s2 p1 -> (32,1,1)
  if (tid < 32) {
    float acc = c2b[tid];
    for (int ic = 0; ic < 16; ic++)
      for (int ky = 0; ky < 4; ky++) {
        int iy = -1 + ky;
        if (iy < 0 || iy >= 2) continue;
        for (int kx = 0; kx < 4; kx++) {
          int ix = -1 + kx;
          if (ix < 0 || ix >= 3) continue;
          acc += h1[ic][iy][ix] * c2w[((tid * 16 + ic) * 4 + ky) * 4 + kx];
        }
      }
    h2[tid] = fmaxf(acc, 0.0f);
  }
  __syncthreads();
  // conv3: (16,32,3,3) s2 p1 -> (16,1,1), only center tap valid
  if (tid < 16) {
    float acc = c3b[tid];
    for (int ic = 0; ic < 32; ic++)
      acc += h2[ic] * c3w[(tid * 32 + ic) * 9 + 4];
    feats[tid] = fmaxf(acc, 0.0f);
  }
  // meta MLP hidden
  if (tid >= 32 && tid < 64) {
    int j = tid - 32;
    float acc = mb1[j];
    const float* mr = meta + (size_t)bt * 4;
    for (int i = 0; i < 4; i++) acc += mr[i] * mw1[i * 32 + j];
    mh[j] = fmaxf(acc, 0.0f);
  }
  __syncthreads();
  if (tid < 4) {
    float acc = mb2[tid];
    for (int i = 0; i < 32; i++) acc += mh[i] * mw2[i * 4 + tid];
    me[tid] = acc;
  }
  __syncthreads();

  float rtg = rtgs[bt];
  int a_id = actions[bt];
  int ts = timesteps[b];
  for (int i = 0; i < 4; i++) {
    int c = tid * 4 + i;
    float se;
    if (c < kC - 4) {
      float acc = encb[c];
      for (int f = 0; f < 16; f++) acc += feats[f] * encw[f * (kC - 4) + c];
      se = acc;
    } else {
      se = me[c - (kC - 4)];
    }
    float st_e = tanhf(se);
    float rt_e = tanhf(rtg * retw[c] + retb[c]);
    float ac_e = actemb[(size_t)a_id * kC + c];
    float gp = gposemb[(size_t)ts * kC + c];
    size_t base = ((size_t)b * kS + 3 * t) * kC + c;
    x[base]           = rt_e + gp + posemb[(size_t)(3 * t + 0) * kC + c];
    x[base + kC]      = st_e + gp + posemb[(size_t)(3 * t + 1) * kC + c];
    x[base + 2 * kC]  = ac_e + gp + posemb[(size_t)(3 * t + 2) * kC + c];
  }
}

// ---------------- layernorm: fp32 row -> bf16 row ----------------
__global__ __launch_bounds__(256) void ln_k(const float* __restrict__ x,
                                            const float* __restrict__ g,
                                            const float* __restrict__ bb,
                                            unsigned short* __restrict__ out) {
  int row = blockIdx.x;
  const float* xr = x + (size_t)row * kC;
  int t = threadIdx.x;
  float v[4], s = 0.f, s2 = 0.f;
#pragma unroll
  for (int i = 0; i < 4; i++) {
    v[i] = xr[t * 4 + i];
    s += v[i];
    s2 += v[i] * v[i];
  }
  __shared__ float sh1[256], sh2[256];
  sh1[t] = s; sh2[t] = s2;
  __syncthreads();
  for (int off = 128; off > 0; off >>= 1) {
    if (t < off) { sh1[t] += sh1[t + off]; sh2[t] += sh2[t + off]; }
    __syncthreads();
  }
  float mean = sh1[0] * (1.0f / kC);
  float var = sh2[0] * (1.0f / kC) - mean * mean;
  float inv = rsqrtf(var + 1e-5f);
#pragma unroll
  for (int i = 0; i < 4; i++) {
    int c = t * 4 + i;
    out[(size_t)row * kC + c] = f2bf((v[i] - mean) * inv * g[c] + bb[c]);
  }
}

// ---------------- WMMA GEMM (LDS double-buffered, async global->LDS staging) ----------------
// out = A[M,K](bf16) * B^T[N,K](bf16) + bias, block tile 128x64, BK=64.
enum { EPI_QK = 0, EPI_VT = 1, EPI_RESID = 2, EPI_GELU = 3, EPI_HEAD = 4 };

template <int EPI>
__global__ __launch_bounds__(256) void gemm_k(const unsigned short* __restrict__ A,
                                              const unsigned short* __restrict__ BT,
                                              const float* __restrict__ bias,
                                              unsigned short* __restrict__ outb,
                                              float* __restrict__ outf,
                                              int N, int K) {
  constexpr int BM = 128, BN = 64, BK = 64;
  constexpr int LDT = BK + 8;                       // 72 elems = 144B row stride
  __shared__ unsigned short As[2][BM * LDT];        // 2 * 18 KB
  __shared__ unsigned short Bs[2][BN * LDT];        // 2 *  9 KB

  int tid = threadIdx.x;
  int lane = tid & 31, w = tid >> 5;
  int wm = w >> 1, wn = w & 1;
  int mb0 = blockIdx.y * BM, nb0 = blockIdx.x * BN;
  int lrow = lane & 15, lhi = lane >> 4;

  // per-thread chunk mapping for staging (16B chunks, 8 chunks per 64-elem row)
  int cr = tid >> 3;            // row within tile (A: +0/+32/+64/+96; B: +0/+32)
  int ck = (tid & 7) * 8;       // k element offset

  auto stage = [&](int buf, int k0) {
#pragma unroll
    for (int i = 0; i < 4; i++) {
      int r = cr + i * 32;
      cp16_g2l(&As[buf][r * LDT + ck], A + (size_t)(mb0 + r) * K + k0 + ck);
    }
#pragma unroll
    for (int i = 0; i < 2; i++) {
      int r = cr + i * 32;
      cp16_g2l(&Bs[buf][r * LDT + ck], BT + (size_t)(nb0 + r) * K + k0 + ck);
    }
  };

  v8f acc[2][2] = {};
  int nK = K / BK;
  stage(0, 0);
  wait_async();
  __syncthreads();

  int buf = 0;
  for (int ks = 0; ks < nK; ks++) {
    if (ks + 1 < nK) stage(buf ^ 1, (ks + 1) * BK);
#pragma unroll
    for (int kk = 0; kk < 2; kk++) {
      const unsigned short* a0 = &As[buf][(wm * 32 + lrow) * LDT + kk * 32 + lhi * 8];
      const unsigned short* a1 = &As[buf][(wm * 32 + 16 + lrow) * LDT + kk * 32 + lhi * 8];
      const unsigned short* b0 = &Bs[buf][(wn * 32 + lrow) * LDT + kk * 32 + lhi * 8];
      const unsigned short* b1 = &Bs[buf][(wn * 32 + 16 + lrow) * LDT + kk * 32 + lhi * 8];
      v16bf af0 = load2x16(a0, a0 + 16);
      v16bf af1 = load2x16(a1, a1 + 16);
      v16bf bf0 = load2x16(b0, b0 + 16);
      v16bf bf1 = load2x16(b1, b1 + 16);
      acc[0][0] = wmma_bf16(af0, bf0, acc[0][0]);
      acc[0][1] = wmma_bf16(af0, bf1, acc[0][1]);
      acc[1][0] = wmma_bf16(af1, bf0, acc[1][0]);
      acc[1][1] = wmma_bf16(af1, bf1, acc[1][1]);
    }
    wait_async();
    __syncthreads();
    buf ^= 1;
  }

  int mb = mb0 + wm * 32;
  int nb = nb0 + wn * 32;
#pragma unroll
  for (int i = 0; i < 2; i++)
#pragma unroll
    for (int j = 0; j < 2; j++) {
      int ncol = nb + j * 16 + lrow;
      float bv = (EPI == EPI_HEAD) ? 0.0f : bias[ncol];
#pragma unroll
      for (int g = 0; g < 8; g++) {
        int mrow = mb + i * 16 + g + 8 * lhi;
        float val = acc[i][j][g] + bv;
        if (EPI == EPI_QK) {
          outb[(size_t)mrow * N + ncol] = f2bf(val);
        } else if (EPI == EPI_VT) {
          int bi = mrow / kS, s = mrow % kS;
          int h = ncol >> 6, d = ncol & 63;
          outb[(((size_t)bi * kH + h) * kD + d) * kS + s] = f2bf(val);
        } else if (EPI == EPI_RESID) {
          outf[(size_t)mrow * N + ncol] += val;
        } else if (EPI == EPI_GELU) {
          float ge = 0.5f * val * (1.0f + erff(val * 0.70710678118f));
          outb[(size_t)mrow * N + ncol] = f2bf(ge);
        } else if (EPI == EPI_HEAD) {
          if (ncol < kA) {
            int bi = mrow / kS, s = mrow % kS;
            if (s % 3 == 1)
              outf[((size_t)bi * kT + (s - 1) / 3) * kA + ncol] = val;
          }
        }
      }
    }
}

// ---------------- attention: one wave per (b, h, 16-query tile) ----------------
__global__ __launch_bounds__(32) void attn_k(const unsigned short* __restrict__ q,
                                             const unsigned short* __restrict__ k,
                                             const unsigned short* __restrict__ vT,
                                             unsigned short* __restrict__ y) {
  int qt = blockIdx.x, h = blockIdx.y, b = blockIdx.z;
  int lane = threadIdx.x;
  int lrow = lane & 15, lhi = lane >> 4;

  __shared__ float sc[16][kS + 8];

  // Q fragments (two 16x32 halves of d=64)
  int sq = qt * 16 + lrow;
  const unsigned short* qrow = q + (size_t)(b * kS + sq) * kC + h * kD;
  v16bf qf[2];
#pragma unroll
  for (int half = 0; half < 2; half++)
    qf[half] = load2x16(qrow + half * 32 + lhi * 8, qrow + half * 32 + lhi * 8 + 16);

  int L = (qt + 1) * 16;

  // pass 1: scores = Q K^T / sqrt(d), causal mask -> LDS
  for (int kt = 0; kt <= qt; kt++) {
    int sk = kt * 16 + lrow;
    const unsigned short* krow = k + (size_t)(b * kS + sk) * kC + h * kD;
    v8f c = {};
#pragma unroll
    for (int half = 0; half < 2; half++) {
      v16bf kf = load2x16(krow + half * 32 + lhi * 8, krow + half * 32 + lhi * 8 + 16);
      c = wmma_bf16(qf[half], kf, c);
    }
    int keyc = kt * 16 + lrow;
#pragma unroll
    for (int g = 0; g < 8; g++) {
      int m = g + 8 * lhi;
      float val = c[g] * 0.125f;           // 1/sqrt(64)
      if (keyc > qt * 16 + m) val = -1e30f;
      sc[m][keyc] = val;
    }
  }
  __syncthreads();

  // softmax per query row
  if (lane < 16) {
    float mx = -3.4e38f;
    for (int j = 0; j < L; j++) mx = fmaxf(mx, sc[lane][j]);
    float sum = 0.f;
    for (int j = 0; j < L; j++) {
      float e = __expf(sc[lane][j] - mx);
      sc[lane][j] = e;
      sum += e;
    }
    float inv = 1.0f / sum;
    for (int j = 0; j < L; j++) sc[lane][j] *= inv;
    for (int j = L; j < L + 16 && j < kS; j++) sc[lane][j] = 0.f;  // pad partial k-tile
  }
  __syncthreads();

  // pass 2: y = P V   (contract over keys in steps of 32)
  v8f acc[4] = {};
  int nk2 = (L + 31) / 32;
  for (int kt2 = 0; kt2 < nk2; kt2++) {
    FragU ua;
    const float* pr = &sc[lrow][kt2 * 32 + lhi * 8];
#pragma unroll
    for (int e = 0; e < 8; e++) ua.us[e] = f2bf(pr[e]);
#pragma unroll
    for (int e = 0; e < 8; e++) ua.us[8 + e] = f2bf(pr[16 + e]);
    v16bf pf = ua.v;
#pragma unroll
    for (int dt = 0; dt < 4; dt++) {
      const unsigned short* vrow =
          vT + (((size_t)b * kH + h) * kD + dt * 16 + lrow) * kS + kt2 * 32 + lhi * 8;
      v16bf vf = load2x16(vrow, vrow + 16);
      acc[dt] = wmma_bf16(pf, vf, acc[dt]);
    }
  }
#pragma unroll
  for (int dt = 0; dt < 4; dt++)
#pragma unroll
    for (int g = 0; g < 8; g++) {
      int m = g + 8 * lhi;
      int srow = qt * 16 + m;
      y[(size_t)(b * kS + srow) * kC + h * kD + dt * 16 + lrow] = f2bf(acc[dt][g]);
    }
}

// ---------------- host driver ----------------
extern "C" void kernel_launch(void* const* d_in, const int* in_sizes, int n_in,
                              void* d_out, int out_size, void* d_ws, size_t ws_size,
                              hipStream_t stream) {
  (void)in_sizes; (void)n_in; (void)out_size; (void)ws_size;
  const float* states   = (const float*)d_in[0];
  const float* rtgs     = (const float*)d_in[1];
  const float* meta     = (const float*)d_in[2];
  // d_in[3] = circuit_feas (unused: zeroed in reference)
  const float* c1w = (const float*)d_in[4];  const float* c1b = (const float*)d_in[5];
  const float* c2w = (const float*)d_in[6];  const float* c2b = (const float*)d_in[7];
  const float* c3w = (const float*)d_in[8];  const float* c3b = (const float*)d_in[9];
  const float* encw = (const float*)d_in[10]; const float* encb = (const float*)d_in[11];
  const float* mw1 = (const float*)d_in[12]; const float* mb1 = (const float*)d_in[13];
  const float* mw2 = (const float*)d_in[14]; const float* mb2 = (const float*)d_in[15];
  const float* retw = (const float*)d_in[16]; const float* retb = (const float*)d_in[17];
  const float* actemb = (const float*)d_in[18];
  const float* posemb = (const float*)d_in[19];
  const float* gposemb = (const float*)d_in[20];
  const float* ln1g = (const float*)d_in[21]; const float* ln1b = (const float*)d_in[22];
  const float* ln2g = (const float*)d_in[23]; const float* ln2b = (const float*)d_in[24];
  const float* Wq = (const float*)d_in[25]; const float* bq = (const float*)d_in[26];
  const float* Wk = (const float*)d_in[27]; const float* bk = (const float*)d_in[28];
  const float* Wv = (const float*)d_in[29]; const float* bv = (const float*)d_in[30];
  const float* Wo = (const float*)d_in[31]; const float* bo = (const float*)d_in[32];
  const float* Wm1 = (const float*)d_in[33]; const float* bm1 = (const float*)d_in[34];
  const float* Wm2 = (const float*)d_in[35]; const float* bm2 = (const float*)d_in[36];
  const float* lnfg = (const float*)d_in[37]; const float* lnfb = (const float*)d_in[38];
  const float* headw = (const float*)d_in[39];
  const int* actions = (const int*)d_in[40];
  const int* timesteps = (const int*)d_in[41];

  char* ws = (char*)d_ws;
  size_t off = 0;
  auto take = [&](size_t bytes) -> void* {
    void* p = ws + off;
    off = (off + bytes + 255) & ~(size_t)255;
    return p;
  };

  float* x            = (float*)take((size_t)kM * kC * 4);
  unsigned short* hx  = (unsigned short*)take((size_t)kM * kC * 2);
  unsigned short* qb  = (unsigned short*)take((size_t)kM * kC * 2);
  unsigned short* kbf = (unsigned short*)take((size_t)kM * kC * 2);
  unsigned short* vT  = (unsigned short*)take((size_t)kB * kH * kD * kS * 2);
  unsigned short* yb  = (unsigned short*)take((size_t)kM * kC * 2);
  unsigned short* ff  = (unsigned short*)take((size_t)kM * kF * 2);
  unsigned short* wqT[kL], *wkT[kL], *wvT[kL], *woT[kL], *wm1T[kL], *wm2T[kL];
  for (int l = 0; l < kL; l++) {
    wqT[l]  = (unsigned short*)take((size_t)kC * kC * 2);
    wkT[l]  = (unsigned short*)take((size_t)kC * kC * 2);
    wvT[l]  = (unsigned short*)take((size_t)kC * kC * 2);
    woT[l]  = (unsigned short*)take((size_t)kC * kC * 2);
    wm1T[l] = (unsigned short*)take((size_t)kF * kC * 2);
    wm2T[l] = (unsigned short*)take((size_t)kC * kF * 2);
  }
  unsigned short* headT = (unsigned short*)take((size_t)128 * kC * 2);

  auto cast = [&](const float* src, unsigned short* dst, int K, int N, int Np) {
    size_t tot = (size_t)Np * K;
    castT_k<<<(unsigned)((tot + 255) / 256), 256, 0, stream>>>(src, dst, K, N, Np);
  };
  for (int l = 0; l < kL; l++) {
    cast(Wq + (size_t)l * kC * kC, wqT[l], kC, kC, kC);
    cast(Wk + (size_t)l * kC * kC, wkT[l], kC, kC, kC);
    cast(Wv + (size_t)l * kC * kC, wvT[l], kC, kC, kC);
    cast(Wo + (size_t)l * kC * kC, woT[l], kC, kC, kC);
    cast(Wm1 + (size_t)l * kC * kF, wm1T[l], kC, kF, kF);
    cast(Wm2 + (size_t)l * kF * kC, wm2T[l], kF, kC, kC);
  }
  cast(headw, headT, kC, kA, 128);

  // embedding -> x (B, 3T, C) fp32
  embed_k<<<kB * kT, 256, 0, stream>>>(states, rtgs, meta, c1w, c1b, c2w, c2b, c3w, c3b,
                                       encw, encb, mw1, mb1, mw2, mb2, retw, retb,
                                       actemb, posemb, gposemb, actions, timesteps, x);

  dim3 gC(kC / 64, kM / 128);   // N=1024 GEMMs
  dim3 gF(kF / 64, kM / 128);   // N=4096 GEMM
  dim3 gA(kS / 16, kH, kB);     // attention
  for (int l = 0; l < kL; l++) {
    ln_k<<<kM, 256, 0, stream>>>(x, ln1g + l * kC, ln1b + l * kC, hx);
    gemm_k<EPI_QK><<<gC, 256, 0, stream>>>(hx, wqT[l], bq + l * kC, qb, nullptr, kC, kC);
    gemm_k<EPI_QK><<<gC, 256, 0, stream>>>(hx, wkT[l], bk + l * kC, kbf, nullptr, kC, kC);
    gemm_k<EPI_VT><<<gC, 256, 0, stream>>>(hx, wvT[l], bv + l * kC, vT, nullptr, kC, kC);
    attn_k<<<gA, 32, 0, stream>>>(qb, kbf, vT, yb);
    gemm_k<EPI_RESID><<<gC, 256, 0, stream>>>(yb, woT[l], bo + l * kC, nullptr, x, kC, kC);
    ln_k<<<kM, 256, 0, stream>>>(x, ln2g + l * kC, ln2b + l * kC, hx);
    gemm_k<EPI_GELU><<<gF, 256, 0, stream>>>(hx, wm1T[l], bm1 + l * kF, ff, nullptr, kF, kC);
    gemm_k<EPI_RESID><<<gC, 256, 0, stream>>>(ff, wm2T[l], bm2 + l * kC, nullptr, x, kC, kF);
  }
  // final LN + head, scattered to state-token outputs
  ln_k<<<kM, 256, 0, stream>>>(x, lnfg, lnfb, hx);
  dim3 gH((kA + 63) / 64, kM / 128);
  gemm_k<EPI_HEAD><<<gH, 256, 0, stream>>>(hx, headT, nullptr, nullptr, (float*)d_out, kA, kC);
}
